// BlockNTPTransformer_40278203301947
// MI455X (gfx1250) — matrix operand order repaired and checked by reading, not compiled, not run on hardware
//
#include <hip/hip_runtime.h>
#include <hip/hip_bf16.h>

// ---------------- problem constants ----------------
#define BB   2
#define TT   1024
#define LL   16
#define DD   512
#define HH   8
#define DH   64
#define DFF  2048
#define VV   32000
#define SS   2048          // 2*T concat sequence
#define NEGV (-1000000000.0f)

// GEMM block tiling
#define BM   128
#define BN   64
#define BK   64
#define LPITCH 80          // LDS row pitch in elements (64 data + 16 pad = 160B)

typedef __bf16 bf16;
typedef __attribute__((ext_vector_type(16))) __bf16 v16bf;
typedef __attribute__((ext_vector_type(8)))  __bf16 v8bf;
typedef __attribute__((ext_vector_type(8)))  float  v8f;
typedef __attribute__((ext_vector_type(4)))  unsigned int u32x4;
typedef __attribute__((ext_vector_type(8)))  int i32x8;
typedef __attribute__((ext_vector_type(4)))  int i32x4;

__device__ __forceinline__ bf16 f2bf(float f) {
    union { float f; unsigned int u; } c; c.f = f;
    unsigned int u = c.u;
    unsigned int r = u + 0x7FFFu + ((u >> 16) & 1u);     // round-to-nearest-even
    union { unsigned short s; bf16 b; } o;
    o.s = (unsigned short)(r >> 16);
    return o.b;
}

// ---------------- Tensor Data Mover: 2D bf16 tile -> LDS (with row padding) --------
// Loads a (th x tw) tile of 2-byte elements from row-major global (stride in elems)
// into LDS at lds_off. TDM pad feature inserts 8 dwords (32B) after every 32 dwords
// (128B = one 64-elem row), giving an LDS row pitch of 160B (LPITCH elems).
__device__ __forceinline__ void tdm_load_tile(unsigned lds_off, const void* gptr,
                                              unsigned tw, unsigned th,
                                              unsigned stride_elems)
{
    unsigned long long ga = (unsigned long long)gptr;
    u32x4 g0;
    g0[0] = 1u;                                            // count=1, user descriptor
    g0[1] = lds_off;                                       // lds_addr (bytes)
    g0[2] = (unsigned)(ga & 0xFFFFFFFFull);                // global_addr[31:0]
    g0[3] = (unsigned)((ga >> 32) & 0x1FFFFFFull) | (2u << 30); // addr[56:32] | type=2

    i32x8 g1;
    // [17:16] data_size=1 (2B)  [20] pad_enable  [24:22] pad_interval=4 (32 dwords)
    // [31:25] pad_amount=7 (8 dwords)
    g1[0] = (int)((1u << 16) | (1u << 20) | (4u << 22) | (7u << 25));
    g1[1] = (int)((tw & 0xFFFFu) << 16);                   // tensor_dim0[15:0] @ bits63:48
    g1[2] = (int)(((tw >> 16) & 0xFFFFu) | ((th & 0xFFFFu) << 16)); // dim0 hi | dim1 lo
    g1[3] = (int)(((th >> 16) & 0xFFFFu) | ((tw & 0xFFFFu) << 16)); // dim1 hi | tile_dim0
    g1[4] = (int)(th & 0xFFFFu);                           // tile_dim1 | tile_dim2=0
    g1[5] = (int)stride_elems;                             // tensor_dim0_stride[31:0]
    g1[6] = 0;                                             // stride hi | dim1_stride lo
    g1[7] = 0;
    i32x4 z4 = {0, 0, 0, 0};
    i32x8 z8 = {0, 0, 0, 0, 0, 0, 0, 0};
    // 6-arg form (clang-23 / therock headers): (g0, g1, g2, g3, g4, cpol)
    __builtin_amdgcn_tensor_load_to_lds(g0, g1, z4, z4, z8, 0);
}

__device__ __forceinline__ unsigned lds_offset_of(const void* p) {
    // generic pointer to LDS: low 32 bits carry the LDS byte offset
    return (unsigned)(unsigned long long)p;
}

// ---------------- bf16 WMMA GEMM, LDS-tiled via TDM, double-buffered ----------------
// C[M,N] (fp32, ldc) = scale * (A[M,K] @ Bt[N,K]^T) + bias + resid, optional GELU.
// A row-major bf16 (lda), Bt transposed: row n holds K contiguous bf16 (ldb).
// Block = 256 threads (8 waves) -> 128x64 tile; wave(4m x 2n) -> 32x32 via 4 WMMAs/kk.
// TDM loads tile i+1 while all waves compute tile i; TENSORcnt<=2 gates the handoff
// (TDM ops complete in order per wave, so <=2 pending => the older pair has landed).
__global__ void __launch_bounds__(256)
gemm_bf16_wmma_k(const bf16* __restrict__ A, int lda,
                 const bf16* __restrict__ Bt, int ldb,
                 float* __restrict__ C, int ldc,
                 int M, int N, int K,
                 const float* __restrict__ bias,
                 const float* __restrict__ resid,
                 float scale, int fuse_gelu)
{
    __shared__ bf16 smA[2][BM * LPITCH];   // 2 x 20 KB
    __shared__ bf16 smB[2][BN * LPITCH];   // 2 x 10 KB

    const int wave = threadIdx.x >> 5;
    const int lane = threadIdx.x & 31;
    const int half = lane >> 4;           // 0: lanes 0-15, 1: lanes 16-31
    const int l15  = lane & 15;
    const int mw   = wave >> 1;           // 0..3
    const int nw   = wave & 1;            // 0..1

    const int blkM = blockIdx.y * BM;
    const int blkN = blockIdx.x * BN;

    const unsigned offA[2] = { lds_offset_of(&smA[0][0]), lds_offset_of(&smA[1][0]) };
    const unsigned offB[2] = { lds_offset_of(&smB[0][0]), lds_offset_of(&smB[1][0]) };
    const bf16* gA = A  + (size_t)blkM * lda;
    const bf16* gB = Bt + (size_t)blkN * ldb;

    v8f acc00 = {}, acc01 = {}, acc10 = {}, acc11 = {};

    // per-lane LDS fragment bases (element offsets), layouts per ISA 7.12.2:
    //  A 16-bit 16x32: lane<16 -> K {0..7,16..23}; lane>=16 -> K {8..15,24..31}
    //  B 32x16: lanes 0-15 K {0..7,16..23} of col n; lanes 16-31 K {8..15,24..31}
    const int arow0 = (mw * 32 + l15) * LPITCH + (half ? 8 : 0);
    const int arow1 = arow0 + 16 * LPITCH;
    const int brow0 = (nw * 32 + l15) * LPITCH + (half ? 8 : 0);
    const int brow1 = brow0 + 16 * LPITCH;

    const int nsteps = K / BK;

    // prologue: stage tile 0 into buffer 0
    if (wave == 0) {
        tdm_load_tile(offA[0], gA, BK, BM, (unsigned)lda);
        tdm_load_tile(offB[0], gB, BK, BN, (unsigned)ldb);
    }

    for (int i = 0; i < nsteps; ++i) {
        const int cur = i & 1;
        __syncthreads();                  // all waves done reading buffer (1-cur)
        if (wave == 0) {
            if (i + 1 < nsteps) {
                const int k1 = (i + 1) * BK;
                tdm_load_tile(offA[1 - cur], gA + k1, BK, BM, (unsigned)lda);
                tdm_load_tile(offB[1 - cur], gB + k1, BK, BN, (unsigned)ldb);
                __builtin_amdgcn_s_wait_tensorcnt(2);   // tile i landed; i+1 in flight
            } else {
                __builtin_amdgcn_s_wait_tensorcnt(0);   // last tile landed
            }
        }
        __syncthreads();                  // tile i visible to all waves

        const bf16* curA = smA[cur];
        const bf16* curB = smB[cur];
#pragma unroll
        for (int kk = 0; kk < BK; kk += 32) {
            v8bf a00 = *(const v8bf*)(curA + arow0 + kk);
            v8bf a01 = *(const v8bf*)(curA + arow0 + kk + 16);
            v16bf av0 = __builtin_shufflevector(a00, a01,
                          0,1,2,3,4,5,6,7,8,9,10,11,12,13,14,15);
            v8bf a10 = *(const v8bf*)(curA + arow1 + kk);
            v8bf a11 = *(const v8bf*)(curA + arow1 + kk + 16);
            v16bf av1 = __builtin_shufflevector(a10, a11,
                          0,1,2,3,4,5,6,7,8,9,10,11,12,13,14,15);
            v16bf bv0 = __builtin_shufflevector(*(const v8bf*)(curB + brow0 + kk), *(const v8bf*)(curB + brow0 + kk + 16),
                          0,1,2,3,4,5,6,7,8,9,10,11,12,13,14,15);
            v16bf bv1 = __builtin_shufflevector(*(const v8bf*)(curB + brow1 + kk), *(const v8bf*)(curB + brow1 + kk + 16),
                          0,1,2,3,4,5,6,7,8,9,10,11,12,13,14,15);

            acc00 = __builtin_amdgcn_wmma_f32_16x16x32_bf16(false, av0, false, bv0, (short)0, acc00, false, false);
            acc01 = __builtin_amdgcn_wmma_f32_16x16x32_bf16(false, av0, false, bv1, (short)0, acc01, false, false);
            acc10 = __builtin_amdgcn_wmma_f32_16x16x32_bf16(false, av1, false, bv0, (short)0, acc10, false, false);
            acc11 = __builtin_amdgcn_wmma_f32_16x16x32_bf16(false, av1, false, bv1, (short)0, acc11, false, false);
        }
    }

    // C/D layout: lane<16 -> (M = m0+i, N = n0+lane); lane>=16 -> (M = m0+8+i, N = n0+lane-16)
    const int mbase = blkM + mw * 32 + (half ? 8 : 0);
    const int nbase = blkN + nw * 32 + l15;
    v8f accs[4] = {acc00, acc01, acc10, acc11};
#pragma unroll
    for (int t = 0; t < 4; ++t) {
        int mi = t >> 1, ni = t & 1;
        int n = nbase + ni * 16;
        float badd = bias ? bias[n] : 0.0f;
#pragma unroll
        for (int i = 0; i < 8; ++i) {
            int m = mbase + mi * 16 + i;
            float v = accs[t][i] * scale + badd;
            if (fuse_gelu) {
                float u = 0.7978845608028654f * (v + 0.044715f * v * v * v);
                v = 0.5f * v * (1.0f + tanhf(u));
            }
            size_t idx = (size_t)m * ldc + n;
            if (resid) v += resid[idx];
            C[idx] = v;
        }
    }
}

// ---------------- embedding + positions ----------------
__global__ void __launch_bounds__(256)
embed_k(const int* __restrict__ tok, const float* __restrict__ emb,
        const float* __restrict__ pos, const float* __restrict__ maskt,
        float* __restrict__ x)
{
    size_t i = (size_t)blockIdx.x * 256 + threadIdx.x;
    if (i >= (size_t)BB * SS * DD) return;
    int d = (int)(i % DD);
    int s = (int)((i / DD) % SS);
    int b = (int)(i / ((size_t)DD * SS));
    float e = (s < TT) ? emb[(size_t)tok[b * TT + s] * DD + d]
                       : maskt[((s - TT) % LL) * DD + d];
    float p = pos[(size_t)((s < TT) ? s : s - TT) * DD + d];
    x[i] = e + p;
}

// ---------------- layernorm -> bf16 (D=512, one row / block of 256) ----------------
__global__ void __launch_bounds__(256)
layernorm_bf16_k(const float* __restrict__ x, const float* __restrict__ s,
                 const float* __restrict__ bsh, bf16* __restrict__ o)
{
    int r = blockIdx.x;
    const float* row = x + (size_t)r * DD;
    __shared__ float r1[256], r2[256];
    float a0 = row[threadIdx.x];
    float a1 = row[threadIdx.x + 256];
    r1[threadIdx.x] = a0 + a1;
    r2[threadIdx.x] = a0 * a0 + a1 * a1;
    __syncthreads();
    for (int st = 128; st > 0; st >>= 1) {
        if (threadIdx.x < st) {
            r1[threadIdx.x] += r1[threadIdx.x + st];
            r2[threadIdx.x] += r2[threadIdx.x + st];
        }
        __syncthreads();
    }
    float mean = r1[0] * (1.0f / DD);
    float var  = r2[0] * (1.0f / DD) - mean * mean;
    float inv  = rsqrtf(var + 1e-5f);
    bf16* orow = o + (size_t)r * DD;
    int c1 = threadIdx.x, c2 = threadIdx.x + 256;
    orow[c1] = f2bf((a0 - mean) * inv * s[c1] + bsh[c1]);
    orow[c2] = f2bf((a1 - mean) * inv * s[c2] + bsh[c2]);
}

// ---------------- fp32 -> bf16 (elementwise) ----------------
__global__ void __launch_bounds__(256)
conv_bf16_k(const float* __restrict__ src, bf16* __restrict__ dst, size_t n)
{
    size_t i = (size_t)blockIdx.x * 256 + threadIdx.x;
    if (i < n) dst[i] = f2bf(src[i]);
}

// ---------------- fp32 [K,N] -> bf16 [N,K] (transpose-convert) ----------------
__global__ void __launch_bounds__(256)
tconv_bf16_k(const float* __restrict__ src, bf16* __restrict__ dst, int K, int N)
{
    size_t i = (size_t)blockIdx.x * 256 + threadIdx.x;
    if (i >= (size_t)K * N) return;
    int n = (int)(i % N);
    int k = (int)(i / N);
    dst[(size_t)n * K + k] = f2bf(src[i]);
}

// ---------------- qkv fp32 [B*S, 3D] -> q,k [B,H,S,DH] bf16, vT [B,H,DH,S] bf16 --------
__global__ void __launch_bounds__(256)
repack_qkv_k(const float* __restrict__ qkv, bf16* __restrict__ q,
             bf16* __restrict__ k, bf16* __restrict__ vT)
{
    size_t i = (size_t)blockIdx.x * 256 + threadIdx.x;
    if (i >= (size_t)BB * SS * HH * DH) return;
    int dh = (int)(i % DH);
    int h  = (int)((i / DH) % HH);
    int s  = (int)((i / ((size_t)DH * HH)) % SS);
    int b  = (int)(i / ((size_t)DH * HH * SS));
    size_t base = ((size_t)b * SS + s) * (3 * DD);
    size_t qi = (((size_t)(b * HH + h)) * SS + s) * DH + dh;
    q[qi] = f2bf(qkv[base + h * DH + dh]);
    k[qi] = f2bf(qkv[base + DD + h * DH + dh]);
    vT[(((size_t)(b * HH + h)) * DH + dh) * SS + s] = f2bf(qkv[base + 2 * DD + h * DH + dh]);
}

// ---------------- masked softmax row -> bf16 probs ----------------
__device__ __forceinline__ bool mask_ok(int r, int c, int masktype)
{
    if (r < TT) return c <= r;
    if (masktype == 0) {                       // block-NTP
        int ch = (r - TT) / LL;
        return (c < ch * LL) || ((c >= TT + ch * LL) && (c < TT + (ch + 1) * LL));
    }
    return (c <= r - TT) || (c == r);          // AR
}

__global__ void __launch_bounds__(256)
softmax_mask_k(const float* __restrict__ sc, bf16* __restrict__ pb, int masktype)
{
    int r = blockIdx.x;
    const float* row = sc + (size_t)r * SS;
    __shared__ float red[256];
    float vals[8];
    float lmax = -3.4e38f;
#pragma unroll
    for (int j = 0; j < 8; ++j) {
        int c = threadIdx.x + 256 * j;
        float v = mask_ok(r, c, masktype) ? row[c] : NEGV;
        vals[j] = v;
        lmax = fmaxf(lmax, v);
    }
    red[threadIdx.x] = lmax; __syncthreads();
    for (int st = 128; st > 0; st >>= 1) {
        if (threadIdx.x < st) red[threadIdx.x] = fmaxf(red[threadIdx.x], red[threadIdx.x + st]);
        __syncthreads();
    }
    float mx = red[0];
    __syncthreads();
    float lsum = 0.0f;
#pragma unroll
    for (int j = 0; j < 8; ++j) { vals[j] = expf(vals[j] - mx); lsum += vals[j]; }
    red[threadIdx.x] = lsum; __syncthreads();
    for (int st = 128; st > 0; st >>= 1) {
        if (threadIdx.x < st) red[threadIdx.x] += red[threadIdx.x + st];
        __syncthreads();
    }
    float inv = 1.0f / red[0];
    bf16* orow = pb + (size_t)r * SS;
#pragma unroll
    for (int j = 0; j < 8; ++j) {
        int c = threadIdx.x + 256 * j;
        orow[c] = f2bf(vals[j] * inv);
    }
}

// ---------------- loss ----------------
__global__ void zero_loss_k(float* p) { *p = 0.0f; }

__global__ void __launch_bounds__(256)
nll_k(const float* __restrict__ logits, const int* __restrict__ tok, float* __restrict__ loss)
{
    int row = blockIdx.x;                 // 0 .. B*(T-1)-1
    int b = row / (TT - 1);
    int j = row % (TT - 1);
    const float* lr = logits + ((size_t)b * SS + TT + j) * VV;
    int label = tok[b * TT + j + 1];
    __shared__ float red[256];
    float lmax = -3.4e38f;
    for (int c = threadIdx.x; c < VV; c += 256) lmax = fmaxf(lmax, lr[c]);
    red[threadIdx.x] = lmax; __syncthreads();
    for (int st = 128; st > 0; st >>= 1) {
        if (threadIdx.x < st) red[threadIdx.x] = fmaxf(red[threadIdx.x], red[threadIdx.x + st]);
        __syncthreads();
    }
    float mx = red[0];
    __syncthreads();
    float lsum = 0.0f;
    for (int c = threadIdx.x; c < VV; c += 256) lsum += expf(lr[c] - mx);
    red[threadIdx.x] = lsum; __syncthreads();
    for (int st = 128; st > 0; st >>= 1) {
        if (threadIdx.x < st) red[threadIdx.x] += red[threadIdx.x + st];
        __syncthreads();
    }
    if (threadIdx.x == 0) {
        float lse = mx + logf(red[0]);
        float nll = lse - lr[label];
        atomicAdd(loss, nll / (float)(BB * (TT - 1)));
    }
}

// ---------------- host orchestration ----------------
static inline void launch_gemm(hipStream_t st,
                               const bf16* A, int lda, const bf16* Bt, int ldb,
                               float* C, int ldc, int M, int N, int K,
                               const float* bias, const float* resid,
                               float scale, int gelu)
{
    dim3 g(N / BN, M / BM);
    gemm_bf16_wmma_k<<<g, 256, 0, st>>>(A, lda, Bt, ldb, C, ldc, M, N, K,
                                        bias, resid, scale, gelu);
}

extern "C" void kernel_launch(void* const* d_in, const int* in_sizes, int n_in,
                              void* d_out, int out_size, void* d_ws, size_t ws_size,
                              hipStream_t stream)
{
    (void)in_sizes; (void)n_in; (void)out_size; (void)ws_size;

    const int*   tok_ids     = (const int*)  d_in[0];
    const float* tok_emb     = (const float*)d_in[1];
    const float* pos_emb     = (const float*)d_in[2];
    const float* mask_tokens = (const float*)d_in[3];
    const float* ln_f_s      = (const float*)d_in[4];
    const float* ln_f_b      = (const float*)d_in[5];
    const float* body[12];
    const float* dec[12];
    for (int i = 0; i < 12; ++i) body[i] = (const float*)d_in[6 + i];
    for (int i = 0; i < 12; ++i) dec[i]  = (const float*)d_in[18 + i];
    // param order: 0 ln1_s 1 ln1_b 2 Wqkv 3 bqkv 4 Wo 5 bo 6 ln2_s 7 ln2_b 8 W1 9 b1 10 W2 11 b2

    float* out_logits = (float*)d_out;
    float* out_loss   = out_logits + (size_t)BB * SS * VV;

    // workspace bump allocator (256B aligned)
    char* w = (char*)d_ws;
    auto alloc = [&](size_t bytes) -> char* {
        char* p = w;
        w += (bytes + 255) & ~(size_t)255;
        return p;
    };
    const size_t MROWS = (size_t)BB * SS;                 // 4096
    float* x        = (float*)alloc(MROWS * DD * 4);
    bf16*  h_bf     = (bf16*) alloc(MROWS * DD * 2);
    float* qkv      = (float*)alloc(MROWS * 3 * DD * 4);
    bf16*  q_bf     = (bf16*) alloc((size_t)BB * HH * SS * DH * 2);
    bf16*  k_bf     = (bf16*) alloc((size_t)BB * HH * SS * DH * 2);
    bf16*  vT_bf    = (bf16*) alloc((size_t)BB * HH * DH * SS * 2);
    float* scores   = (float*)alloc((size_t)SS * SS * 4);
    bf16*  p_bf     = (bf16*) alloc((size_t)SS * SS * 2);
    float* attnO    = (float*)alloc(MROWS * DD * 4);
    bf16*  attnO_bf = (bf16*) alloc(MROWS * DD * 2);
    float* ffn1     = (float*)alloc(MROWS * DFF * 4);
    bf16*  ffn1_bf  = (bf16*) alloc(MROWS * DFF * 2);
    bf16*  wqkv_t   = (bf16*) alloc((size_t)3 * DD * DD * 2);
    bf16*  wo_t     = (bf16*) alloc((size_t)DD * DD * 2);
    bf16*  w1_t     = (bf16*) alloc((size_t)DFF * DD * 2);
    bf16*  w2_t     = (bf16*) alloc((size_t)DD * DFF * 2);
    bf16*  emb_bf   = (bf16*) alloc((size_t)VV * DD * 2);

    auto gblk = [](size_t n) { return (unsigned)((n + 255) / 256); };

    // tok_emb -> bf16 [V, D] (serves as Bt for the logits GEMM: row v holds K=D contiguous)
    conv_bf16_k<<<gblk((size_t)VV * DD), 256, 0, stream>>>(tok_emb, emb_bf, (size_t)VV * DD);

    // embeddings + positions -> x [B,S,D]
    embed_k<<<gblk(MROWS * DD), 256, 0, stream>>>(tok_ids, tok_emb, pos_emb, mask_tokens, x);

    for (int layer = 0; layer < 6; ++layer) {
        const float* const* P = (layer < 4) ? body : dec;
        int li = (layer < 4) ? layer : layer - 4;
        int masktype = (layer < 4) ? 0 : 1;

        const float* ln1_s = P[0]  + (size_t)li * DD;
        const float* ln1_b = P[1]  + (size_t)li * DD;
        const float* Wqkv  = P[2]  + (size_t)li * DD * 3 * DD;
        const float* bqkv  = P[3]  + (size_t)li * 3 * DD;
        const float* Wo    = P[4]  + (size_t)li * DD * DD;
        const float* bo    = P[5]  + (size_t)li * DD;
        const float* ln2_s = P[6]  + (size_t)li * DD;
        const float* ln2_b = P[7]  + (size_t)li * DD;
        const float* W1    = P[8]  + (size_t)li * DD * DFF;
        const float* b1    = P[9]  + (size_t)li * DFF;
        const float* W2    = P[10] + (size_t)li * DFF * DD;
        const float* b2    = P[11] + (size_t)li * DD;

        // LN1 -> bf16
        layernorm_bf16_k<<<(unsigned)MROWS, 256, 0, stream>>>(x, ln1_s, ln1_b, h_bf);

        // qkv = h @ Wqkv + bqkv
        tconv_bf16_k<<<gblk((size_t)DD * 3 * DD), 256, 0, stream>>>(Wqkv, wqkv_t, DD, 3 * DD);
        launch_gemm(stream, h_bf, DD, wqkv_t, DD, qkv, 3 * DD,
                    (int)MROWS, 3 * DD, DD, bqkv, nullptr, 1.0f, 0);

        // split heads, v transposed
        repack_qkv_k<<<gblk((size_t)BB * SS * HH * DH), 256, 0, stream>>>(qkv, q_bf, k_bf, vT_bf);

        // attention per (b,h): scores = Q K^T / 8 ; softmax+mask ; O = P V
        for (int bh = 0; bh < BB * HH; ++bh) {
            int b = bh / HH, h = bh % HH;
            const bf16* qb = q_bf  + (size_t)bh * SS * DH;
            const bf16* kb = k_bf  + (size_t)bh * SS * DH;
            const bf16* vb = vT_bf + (size_t)bh * DH * SS;
            launch_gemm(stream, qb, DH, kb, DH, scores, SS,
                        SS, SS, DH, nullptr, nullptr, 0.125f, 0);
            softmax_mask_k<<<SS, 256, 0, stream>>>(scores, p_bf, masktype);
            float* ob = attnO + (size_t)b * SS * DD + (size_t)h * DH;
            launch_gemm(stream, p_bf, SS, vb, SS, ob, DD,
                        SS, DH, SS, nullptr, nullptr, 1.0f, 0);
        }

        // x = x + attnO @ Wo + bo
        conv_bf16_k<<<gblk(MROWS * DD), 256, 0, stream>>>(attnO, attnO_bf, MROWS * DD);
        tconv_bf16_k<<<gblk((size_t)DD * DD), 256, 0, stream>>>(Wo, wo_t, DD, DD);
        launch_gemm(stream, attnO_bf, DD, wo_t, DD, x, DD,
                    (int)MROWS, DD, DD, bo, x, 1.0f, 0);

        // FFN: x = x + GELU(LN2(x) @ W1 + b1) @ W2 + b2
        layernorm_bf16_k<<<(unsigned)MROWS, 256, 0, stream>>>(x, ln2_s, ln2_b, h_bf);
        tconv_bf16_k<<<gblk((size_t)DD * DFF), 256, 0, stream>>>(W1, w1_t, DD, DFF);
        launch_gemm(stream, h_bf, DD, w1_t, DD, ffn1, DFF,
                    (int)MROWS, DFF, DD, b1, nullptr, 1.0f, 1);
        conv_bf16_k<<<gblk(MROWS * DFF), 256, 0, stream>>>(ffn1, ffn1_bf, MROWS * DFF);
        tconv_bf16_k<<<gblk((size_t)DFF * DD), 256, 0, stream>>>(W2, w2_t, DFF, DD);
        launch_gemm(stream, ffn1_bf, DFF, w2_t, DFF, x, DD,
                    (int)MROWS, DD, DFF, b2, x, 1.0f, 0);
    }

    // final LN then logits = xf @ tok_emb^T  (Bt = emb_bf [V, D])
    layernorm_bf16_k<<<(unsigned)MROWS, 256, 0, stream>>>(x, ln_f_s, ln_f_b, h_bf);
    launch_gemm(stream, h_bf, DD, emb_bf, DD, out_logits, VV,
                (int)MROWS, VV, DD, nullptr, nullptr, 1.0f, 0);

    // loss = mean NLL over logits[:, T:-1, :] vs tok_ids[:, 1:]
    zero_loss_k<<<1, 1, 0, stream>>>(out_loss);
    nll_k<<<BB * (TT - 1), 256, 0, stream>>>(out_logits, tok_ids, out_loss);
}
